// ResLSTMLayers_67413806678544
// MI455X (gfx1250) — compile-verified, matrix operand order checked by reading, not run on hardware
//
#include <hip/hip_runtime.h>
#include <hip/hip_bf16.h>

// Problem constants (from reference)
#define TT   512
#define BB   128
#define INP  1024
#define HH   1024
#define LL   4
#define KK   1024          // == INP == HH
#define NG   (3*HH)        // 3072 gate columns
#define BH   (BB*HH)       // 131072
#define LBH  (LL*BB*HH)    // 524288
#define GSZ  (BB*NG)       // 393216

typedef __attribute__((ext_vector_type(16))) __bf16 v16bf;
typedef __attribute__((ext_vector_type(8)))  __bf16 v8bf;
typedef __attribute__((ext_vector_type(8)))  float  v8f;

// ---------------------------------------------------------------------------
// Load one 16x16x32 bf16 WMMA operand fragment for this lane.
// Both A ([M,K] row-major) and B (W stored [N,K] row-major, i.e. B^T) use the
// same per-lane pattern: 8 contiguous bf16 at (row, kb+koff) and 8 more at
// (row, kb+koff+16), koff = (lane<16 ? 0 : 8).  16-byte aligned (K=1024).
// ---------------------------------------------------------------------------
__device__ __forceinline__ v16bf load_frag(const __bf16* rowptr, int kb, int koff) {
    const v8bf lo = *(const v8bf*)(rowptr + kb + koff);
    const v8bf hi = *(const v8bf*)(rowptr + kb + koff + 16);
    v16bf r;
#pragma unroll
    for (int i = 0; i < 8; ++i) { r[i] = lo[i]; r[i + 8] = hi[i]; }
    return r;
}

// K=1024 inner product accumulated into a 32x64 wave tile (2 M-frags x 4 N-frags)
__device__ __forceinline__ void gemm_k1024(const __bf16* __restrict__ A,
                                           const __bf16* __restrict__ W,
                                           const int am[2], const int wn[4],
                                           int koff, v8f acc[2][4]) {
    const __bf16* arow[2];
    const __bf16* wrow[4];
#pragma unroll
    for (int mf = 0; mf < 2; ++mf) arow[mf] = A + (size_t)am[mf] * KK;
#pragma unroll
    for (int nf = 0; nf < 4; ++nf) wrow[nf] = W + (size_t)wn[nf] * KK;

#pragma unroll 2
    for (int kb = 0; kb < KK; kb += 32) {
        if ((kb & 255) == 0) {
            // keep weight stream hot (global_prefetch_b8)
            __builtin_prefetch(wrow[0] + kb + 512, 0, 0);
            __builtin_prefetch(wrow[2] + kb + 512, 0, 0);
        }
        v16bf af[2], bfr[4];
#pragma unroll
        for (int mf = 0; mf < 2; ++mf) af[mf] = load_frag(arow[mf], kb, koff);
#pragma unroll
        for (int nf = 0; nf < 4; ++nf) bfr[nf] = load_frag(wrow[nf], kb, koff);
#pragma unroll
        for (int mf = 0; mf < 2; ++mf)
#pragma unroll
            for (int nf = 0; nf < 4; ++nf)
                acc[mf][nf] = __builtin_amdgcn_wmma_f32_16x16x32_bf16(
                    false, af[mf], false, bfr[nf], (short)0, acc[mf][nf],
                    false, false);
    }
}

// Store 32x64 wave tile.  C layout: lane n = lane&15, row m = vgpr + (lane>=16)*8.
__device__ __forceinline__ void store_acc(float* __restrict__ C, int ldc,
                                          int m0, int n0, int lane,
                                          v8f acc[2][4]) {
    const int n  = lane & 15;
    const int mo = (lane >> 4) << 3;
#pragma unroll
    for (int mf = 0; mf < 2; ++mf)
#pragma unroll
        for (int nf = 0; nf < 4; ++nf)
#pragma unroll
            for (int r = 0; r < 8; ++r)
                C[(size_t)(m0 + mf * 16 + mo + r) * ldc + (n0 + nf * 16 + n)] =
                    acc[mf][nf][r];
}

// ---------------------------------------------------------------------------
// Per-(t,layer) GEMM, fully K-parallel across blocks: 88 blocks x 256 threads.
// Each gate contribution goes to its own partial buffer so every block runs a
// single K=1024 pass (32 WMMA iterations on the recurrence critical path):
//  blocks  0..23: g0 = out@Wii^T    (M=128,N=3072)
//  blocks 24..47: g1 = h  @Wih^T    (M=128,N=3072)
//  blocks 48..71: g2 = c  @Wic^T    (M=128,N=3072)
//  blocks 72..79: cellpre = h@Whh^T (M=128,N=1024)
//  blocks 80..87: res     = out@Wir^T (M=128,N=1024)
// ---------------------------------------------------------------------------
__global__ void __launch_bounds__(256)
res_lstm_gemm(const __bf16* __restrict__ Aout, const __bf16* __restrict__ Ah,
              const __bf16* __restrict__ Ac,
              const __bf16* __restrict__ Wii, const __bf16* __restrict__ Wih,
              const __bf16* __restrict__ Wic,
              const __bf16* __restrict__ Whh, const __bf16* __restrict__ Wir,
              float* __restrict__ g0, float* __restrict__ g1,
              float* __restrict__ g2, float* __restrict__ cellpre,
              float* __restrict__ res) {
    const int bid  = blockIdx.x;
    const int tid  = threadIdx.x;
    const int lane = tid & 31;
    const int wave = tid >> 5;
    const int wm   = wave & 3;   // 0..3  -> 32-row slab
    const int wn   = wave >> 2;  // 0..1  -> 64-col slab
    const int m0   = wm * 32;
    const int n0w  = wn * 64;
    const int koff = (lane >> 4) << 3;
    const int lr   = lane & 15;

    const __bf16* A;
    const __bf16* W;
    float* C;
    int ldc, nblk;
    if (bid < 24)      { A = Aout; W = Wii; C = g0;      ldc = NG; nblk = bid * 128; }
    else if (bid < 48) { A = Ah;   W = Wih; C = g1;      ldc = NG; nblk = (bid - 24) * 128; }
    else if (bid < 72) { A = Ac;   W = Wic; C = g2;      ldc = NG; nblk = (bid - 48) * 128; }
    else if (bid < 80) { A = Ah;   W = Whh; C = cellpre; ldc = HH; nblk = (bid - 72) * 128; }
    else               { A = Aout; W = Wir; C = res;     ldc = HH; nblk = (bid - 80) * 128; }

    v8f acc[2][4] = {};
    int am[2];
#pragma unroll
    for (int mf = 0; mf < 2; ++mf) am[mf] = m0 + mf * 16 + lr;
    int wnr[4];
#pragma unroll
    for (int nf = 0; nf < 4; ++nf) wnr[nf] = nblk + n0w + nf * 16 + lr;

    gemm_k1024(A, W, am, wnr, koff, acc);
    store_acc(C, ldc, m0, nblk + n0w, lane, acc);
}

__device__ __forceinline__ float sigmoidf_(float x) {
    return 1.0f / (1.0f + __expf(-x));
}

// Elementwise gating: sums the three gate partials, applies biases and
// sigmoid/tanh, updates f32 state and bf16 copies for the next WMMA round.
__global__ void __launch_bounds__(256)
res_lstm_pointwise(const float* __restrict__ g0, const float* __restrict__ g1,
                   const float* __restrict__ g2,
                   const float* __restrict__ cellpre, const float* __restrict__ res,
                   const float* __restrict__ bii, const float* __restrict__ bih,
                   const float* __restrict__ bic, const float* __restrict__ bhh,
                   float* __restrict__ h_state, float* __restrict__ c_state,
                   __bf16* __restrict__ h_bf, __bf16* __restrict__ c_bf,
                   float* __restrict__ out_opt) {
    const int idx = blockIdx.x * blockDim.x + threadIdx.x;   // < B*H
    const int b = idx >> 10;
    const int n = idx & 1023;
    const size_t gr = (size_t)b * NG;
    const float ig = g0[gr + n] + g1[gr + n] + g2[gr + n]
                   + bii[n] + bih[n] + bic[n];
    const float fg = g0[gr + HH + n] + g1[gr + HH + n] + g2[gr + HH + n]
                   + bii[HH + n] + bih[HH + n] + bic[HH + n];
    const float og = g0[gr + 2 * HH + n] + g1[gr + 2 * HH + n] + g2[gr + 2 * HH + n]
                   + bii[2 * HH + n] + bih[2 * HH + n] + bic[2 * HH + n];
    const float cell = tanhf(cellpre[idx] + bhh[n]);
    const float cold = c_state[idx];
    const float cy = sigmoidf_(fg) * cold + sigmoidf_(ig) * cell;
    const float hy = sigmoidf_(og) * (tanhf(cy) + res[idx]);
    c_state[idx] = cy;
    h_state[idx] = hy;
    c_bf[idx] = (__bf16)cy;
    h_bf[idx] = (__bf16)hy;
    if (out_opt) out_opt[idx] = hy;
}

__global__ void cvt_f32_bf16(const float* __restrict__ in, __bf16* __restrict__ out,
                             long long n) {
    long long i = (long long)blockIdx.x * blockDim.x + threadIdx.x;
    const long long stride = (long long)gridDim.x * blockDim.x;
    for (; i < n; i += stride) out[i] = (__bf16)in[i];
}

__global__ void init_state(const float* __restrict__ hidden,
                           float* __restrict__ h_state, float* __restrict__ c_state,
                           __bf16* __restrict__ h_bf, __bf16* __restrict__ c_bf) {
    int i = blockIdx.x * blockDim.x + threadIdx.x;
    const int stride = gridDim.x * blockDim.x;
    for (; i < LBH; i += stride) {
        const float h = hidden[i];
        const float c = hidden[LBH + i];
        h_state[i] = h; c_state[i] = c;
        h_bf[i] = (__bf16)h; c_bf[i] = (__bf16)c;
    }
}

__global__ void final_state_copy(const float* __restrict__ h_state,
                                 const float* __restrict__ c_state,
                                 float* __restrict__ dst) {
    int i = blockIdx.x * blockDim.x + threadIdx.x;
    const int stride = gridDim.x * blockDim.x;
    for (; i < LBH; i += stride) {
        dst[i] = h_state[i];
        dst[LBH + i] = c_state[i];
    }
}

// ---------------------------------------------------------------------------
extern "C" void kernel_launch(void* const* d_in, const int* in_sizes, int n_in,
                              void* d_out, int out_size, void* d_ws, size_t ws_size,
                              hipStream_t stream) {
    // input order: x, hidden, Wii, Wic, Wih, bii, bic, bih, Whh, bhh, Wir
    const float* x      = (const float*)d_in[0];
    const float* hidden = (const float*)d_in[1];
    const float* Wii    = (const float*)d_in[2];
    const float* Wic    = (const float*)d_in[3];
    const float* Wih    = (const float*)d_in[4];
    const float* bii    = (const float*)d_in[5];
    const float* bic    = (const float*)d_in[6];
    const float* bih    = (const float*)d_in[7];
    const float* Whh    = (const float*)d_in[8];
    const float* bhh    = (const float*)d_in[9];
    const float* Wir    = (const float*)d_in[10];
    float* out = (float*)d_out;

    // workspace carve-out (256B aligned)
    char* base = (char*)d_ws;
    size_t off = 0;
    auto alloc = [&](size_t bytes) -> char* {
        char* p = base + off;
        off = (off + bytes + 255) & ~(size_t)255;
        return p;
    };
    const size_t SZ_W3 = (size_t)LL * NG * KK;   // 12.58M elems
    const size_t SZ_W1 = (size_t)LL * HH * KK;   //  4.19M elems

    __bf16* wii_bf = (__bf16*)alloc(SZ_W3 * 2);
    __bf16* wih_bf = (__bf16*)alloc(SZ_W3 * 2);
    __bf16* wic_bf = (__bf16*)alloc(SZ_W3 * 2);
    __bf16* whh_bf = (__bf16*)alloc(SZ_W1 * 2);
    __bf16* wir_bf = (__bf16*)alloc(SZ_W1 * 2);
    float*  h_st   = (float*)alloc((size_t)LBH * 4);
    float*  c_st   = (float*)alloc((size_t)LBH * 4);
    __bf16* h_bf   = (__bf16*)alloc((size_t)LBH * 2);
    __bf16* c_bf   = (__bf16*)alloc((size_t)LBH * 2);
    __bf16* xbf1   = (__bf16*)alloc((size_t)BB * INP * 2);   // per-step fallback
    float*  g0     = (float*)alloc((size_t)GSZ * 4);
    float*  g1     = (float*)alloc((size_t)GSZ * 4);
    float*  g2     = (float*)alloc((size_t)GSZ * 4);
    float*  cellp  = (float*)alloc((size_t)BH * 4);
    float*  resb   = (float*)alloc((size_t)BH * 4);
    // Optional full-sequence bf16 copy of x (removes 512 serial converts from
    // the recurrence chain).  ws_size is fixed across calls -> deterministic.
    __bf16* xbf_all = (__bf16*)(base + off);
    const size_t need_full_x = off + (size_t)TT * BB * INP * 2;
    const bool full_x = (ws_size >= need_full_x);
    (void)in_sizes; (void)n_in; (void)out_size;

    // 1) weights -> bf16 (stay resident in 192MB L2 across the recurrence)
    cvt_f32_bf16<<<2048, 256, 0, stream>>>(Wii, wii_bf, (long long)SZ_W3);
    cvt_f32_bf16<<<2048, 256, 0, stream>>>(Wih, wih_bf, (long long)SZ_W3);
    cvt_f32_bf16<<<2048, 256, 0, stream>>>(Wic, wic_bf, (long long)SZ_W3);
    cvt_f32_bf16<<<1024, 256, 0, stream>>>(Whh, whh_bf, (long long)SZ_W1);
    cvt_f32_bf16<<<1024, 256, 0, stream>>>(Wir, wir_bf, (long long)SZ_W1);

    // 2) initial state; bulk x conversion if workspace allows
    init_state<<<1024, 256, 0, stream>>>(hidden, h_st, c_st, h_bf, c_bf);
    if (full_x)
        cvt_f32_bf16<<<4096, 256, 0, stream>>>(x, xbf_all,
                                               (long long)TT * BB * INP);

    // 3) recurrence: 512 steps x 4 layers, each = K-parallel WMMA GEMM + pointwise
    for (int t = 0; t < TT; ++t) {
        const __bf16* xt;
        if (full_x) {
            xt = xbf_all + (size_t)t * BB * INP;
        } else {
            cvt_f32_bf16<<<512, 256, 0, stream>>>(x + (size_t)t * BB * INP, xbf1,
                                                  (long long)BB * INP);
            xt = xbf1;
        }
        for (int j = 0; j < LL; ++j) {
            const __bf16* Aout = (j == 0) ? xt : (h_bf + (size_t)(j - 1) * BH);
            res_lstm_gemm<<<88, 256, 0, stream>>>(
                Aout, h_bf + (size_t)j * BH, c_bf + (size_t)j * BH,
                wii_bf + (size_t)j * NG * KK, wih_bf + (size_t)j * NG * KK,
                wic_bf + (size_t)j * NG * KK,
                whh_bf + (size_t)j * HH * KK, wir_bf + (size_t)j * HH * KK,
                g0, g1, g2, cellp, resb);
            res_lstm_pointwise<<<BH / 256, 256, 0, stream>>>(
                g0, g1, g2, cellp, resb,
                bii + (size_t)j * NG, bih + (size_t)j * NG, bic + (size_t)j * NG,
                bhh + (size_t)j * HH,
                h_st + (size_t)j * BH, c_st + (size_t)j * BH,
                h_bf + (size_t)j * BH, c_bf + (size_t)j * BH,
                (j == LL - 1) ? (out + (size_t)t * BH) : nullptr);
        }
    }

    // 4) tuple tail: hF then cF
    final_state_copy<<<1024, 256, 0, stream>>>(h_st, c_st, out + (size_t)TT * BH);
}
